// SoftAlignmentModule_18262200943081
// MI455X (gfx1250) — matrix operand order repaired and checked by reading, not compiled, not run on hardware
//
#include <hip/hip_runtime.h>

#define B_    16
#define PL    1024
#define HL    1024
#define D_    256
#define BM    16
#define NEGBIG 1e30f

typedef __attribute__((ext_vector_type(4)))  float  v4f;
typedef __attribute__((ext_vector_type(8)))  float  v8f;
typedef __attribute__((ext_vector_type(4)))  __bf16 v4bf;
typedef __attribute__((ext_vector_type(8)))  __bf16 v8bf;
typedef __attribute__((ext_vector_type(16))) __bf16 v16bf;

// ---- helpers -------------------------------------------------------------

__device__ __forceinline__ v16bf cat8(v8bf lo, v8bf hi) {
  return __builtin_shufflevector(lo, hi, 0,1,2,3,4,5,6,7,8,9,10,11,12,13,14,15);
}

__device__ __forceinline__ v16bf cvt16(v4f a, v4f b, v4f c, v4f d) {
  v16bf r;
#pragma unroll
  for (int j = 0; j < 4; ++j) {
    r[j]      = (__bf16)a[j];
    r[4 + j]  = (__bf16)b[j];
    r[8 + j]  = (__bf16)c[j];
    r[12 + j] = (__bf16)d[j];
  }
  return r;
}

__device__ __forceinline__ v8f wmma_bf16(v16bf a, v16bf b, v8f c) {
  // D(f32 16x16) = A(bf16 16x32) * B(bf16 32x16) + C
  return __builtin_amdgcn_wmma_f32_16x16x32_bf16(false, a, false, b, (short)0, c,
                                                 false, false);
}

// order-preserving float->uint key for atomic max
__device__ __forceinline__ unsigned fkey(float f) {
  unsigned b = __float_as_uint(f);
  return (b & 0x80000000u) ? ~b : (b | 0x80000000u);
}
__device__ __forceinline__ float fdec(unsigned k) {
  unsigned b = (k & 0x80000000u) ? (k & 0x7FFFFFFFu) : ~k;
  return __uint_as_float(b);
}

// ---- prep kernels --------------------------------------------------------

// Wt[n][k] = bf16(W[k][n])   (256x256)
__global__ void k0_wt(const float* __restrict__ W, unsigned short* __restrict__ wt_) {
  int n = blockIdx.x, k = threadIdx.x;
  __bf16* Wt = (__bf16*)wt_;
  Wt[n * D_ + k] = (__bf16)W[k * D_ + n];
}

// Ht[b][d][q] = bf16(H[b][q][d])  (tiled transpose through LDS)
__global__ void k0_ht(const float* __restrict__ H, unsigned short* __restrict__ ht_) {
  __shared__ float tile[16][17];
  int b = blockIdx.z;
  int tx = threadIdx.x & 15, ty = threadIdx.x >> 4;
  int q = blockIdx.x * 16 + ty, d = blockIdx.y * 16 + tx;
  tile[ty][tx] = H[((size_t)(b * HL + q)) * D_ + d];
  __syncthreads();
  __bf16* Ht = (__bf16*)ht_;
  int dd = blockIdx.y * 16 + ty, qq = blockIdx.x * 16 + tx;
  Ht[((size_t)(b * D_ + dd)) * HL + qq] = (__bf16)tile[tx][ty];
}

// Hbf = bf16(H), same [b][q][d] layout; vectorized 4 elems/thread
__global__ void k0_hbf(const float* __restrict__ H, unsigned short* __restrict__ hbf_) {
  size_t i = ((size_t)blockIdx.x * 256 + threadIdx.x) * 4;
  v4f v = *(const v4f*)(H + i);
  v4bf o;
#pragma unroll
  for (int j = 0; j < 4; ++j) o[j] = (__bf16)v[j];
  *(v4bf*)((__bf16*)hbf_ + i) = o;
}

// init colmax keys to fkey(-inf) = ~bits(-inf) = 0x007FFFFF
__global__ void k0_init(unsigned* __restrict__ keys) {
  keys[blockIdx.x * 256 + threadIdx.x] = 0x007FFFFFu;
}

// ---- fused main kernel ---------------------------------------------------
// One workgroup = one (batch, 16-row p block). 256 threads = 8 wave32.
// LDS: sPW    bf16[16][256]  @      0 (  8192 B)
//      sS     f32 [16][1024] @   8192 ( 65536 B)
//      sProb  bf16[16][1024] @  73728 ( 32768 B)
//      sHm    int [1024]     @ 106496 (  4096 B)
//      sPm    int [16]       @ 110592 (    64 B)
//      sRowMax f32[16]       @ 110656 (    64 B)
//      sRowRs  f32[16]       @ 110720 (    64 B)
//      sRed2   f32[16][16]   @ 110784 (  1024 B)
#define SMEM_K2 111808

__global__ __launch_bounds__(256) void k2_fused(
    const float* __restrict__ P,
    const int* __restrict__ pm, const int* __restrict__ hm,
    const float* __restrict__ bias,
    const unsigned short* __restrict__ wt_, const unsigned short* __restrict__ ht_,
    const unsigned short* __restrict__ hbf_,
    unsigned* __restrict__ keys, float* __restrict__ outHyp) {
  extern __shared__ char smem[];
  __bf16* sPW     = (__bf16*)(smem);
  float*  sS      = (float*)(smem + 8192);
  __bf16* sProb   = (__bf16*)(smem + 73728);
  int*    sHm     = (int*)(smem + 106496);
  int*    sPm     = (int*)(smem + 110592);
  float*  sRowMax = (float*)(smem + 110656);
  float*  sRowRs  = (float*)(smem + 110720);
  float*  sRed2   = (float*)(smem + 110784);

  const int tid   = threadIdx.x;
  const int wave  = tid >> 5;
  const int lane  = tid & 31;
  const int lhalf = lane >> 4;   // 0: lanes 0-15, 1: lanes 16-31
  const int lmod  = lane & 15;
  const int b  = blockIdx.y;
  const int p0 = blockIdx.x * BM;
  const __bf16* Wt  = (const __bf16*)wt_;
  const __bf16* Ht  = (const __bf16*)ht_;
  const __bf16* Hbf = (const __bf16*)hbf_;
  const float bv = bias[0];

  // preload masks
  for (int i = tid; i < HL; i += 256) sHm[i] = hm[b * HL + i];
  if (tid < BM) sPm[tid] = pm[b * PL + p0 + tid];

  // ---- Phase A: PW(16x256) = P_block(16x256) @ W, bf16 result into LDS
#pragma unroll
  for (int t2 = 0; t2 < 2; ++t2) {
    const int nt = wave * 2 + t2;           // 16 n-tiles over d
    v8f acc = {};
#pragma unroll
    for (int kt = 0; kt < 8; ++kt) {        // K = 256, 32 per wmma
      const int kb = kt * 32;
      // A frag: row m = lmod of P block; K chunks kb+8*lhalf and +16
      const float* arow = P + ((size_t)(b * PL + p0 + lmod)) * D_ + kb + 8 * lhalf;
      v4f a0 = *(const v4f*)(arow);
      v4f a1 = *(const v4f*)(arow + 4);
      v4f a2 = *(const v4f*)(arow + 16);
      v4f a3 = *(const v4f*)(arow + 20);
      v16bf A = cvt16(a0, a1, a2, a3);
      // B frag: column n = lmod of tile nt; K chunks kb+8*lhalf and +16 from Wt row
      const __bf16* brow = Wt + (nt * 16 + lmod) * D_ + kb + 8 * lhalf;
      v16bf Bf = cat8(*(const v8bf*)brow, *(const v8bf*)(brow + 16));
      acc = wmma_bf16(A, Bf, acc);
    }
#pragma unroll
    for (int r = 0; r < 8; ++r) {
      const int M = r + 8 * lhalf;
      sPW[M * D_ + nt * 16 + lmod] = (__bf16)acc[r];
    }
  }
  __syncthreads();

  // ---- Phase B: S(16x1024) = PW @ Hbf^T (+bias, mask) into LDS f32
  for (int qi = 0; qi < 8; ++qi) {
    const int qt  = wave + qi * 8;          // 64 q-tiles
    const int q0t = qt * 16;
    v8f acc = {};
#pragma unroll
    for (int kt = 0; kt < 8; ++kt) {        // K = 256 (hidden)
      const int kb = kt * 32;
      const __bf16* arow = sPW + lmod * D_ + kb + 8 * lhalf;
      v16bf A = cat8(*(const v8bf*)arow, *(const v8bf*)(arow + 16));
      const __bf16* brow = Hbf + ((size_t)(b * HL + q0t + lmod)) * D_ + kb + 8 * lhalf;
      v16bf Bf = cat8(*(const v8bf*)brow, *(const v8bf*)(brow + 16));
      acc = wmma_bf16(A, Bf, acc);
    }
    const int qg  = q0t + lmod;
    const int hmv = sHm[qg];
#pragma unroll
    for (int r = 0; r < 8; ++r) {
      const int M = r + 8 * lhalf;
      float s = acc[r] + bv;
      if (sPm[M] == 0 || hmv == 0) s -= NEGBIG;
      sS[M * HL + qg] = s;
    }
  }
  __syncthreads();

  // ---- Phase C1: column max -> global atomic max
  for (int q = tid; q < HL; q += 256) {
    float v = -__builtin_inff();
#pragma unroll
    for (int m = 0; m < BM; ++m) v = fmaxf(v, sS[m * HL + q]);
    atomicMax(&keys[b * HL + q], fkey(v));
  }
  // ---- Phase C2: row max (16 threads per row, 64 elems each)
  {
    const int m = tid >> 4, part = tid & 15;
    float lm = -__builtin_inff();
    const int qb = part * 64;
    for (int q = qb; q < qb + 64; ++q) lm = fmaxf(lm, sS[m * HL + q]);
    sRed2[m * 16 + part] = lm;
  }
  __syncthreads();
  if (tid < BM) {
    float mx = -__builtin_inff();
#pragma unroll
    for (int j = 0; j < 16; ++j) mx = fmaxf(mx, sRed2[tid * 16 + j]);
    sRowMax[tid] = mx;
  }
  __syncthreads();
  // ---- Phase C3: row sum of exp
  {
    const int m = tid >> 4, part = tid & 15;
    const float mx = sRowMax[m];
    float ls = 0.f;
    const int qb = part * 64;
    for (int q = qb; q < qb + 64; ++q) ls += __expf(sS[m * HL + q] - mx);
    sRed2[m * 16 + part] = ls;
  }
  __syncthreads();
  if (tid < BM) {
    float sum = 0.f;
#pragma unroll
    for (int j = 0; j < 16; ++j) sum += sRed2[tid * 16 + j];
    sRowRs[tid] = 1.0f / sum;
  }
  __syncthreads();
  // ---- Phase C4: normalized probs -> bf16 LDS
  for (int i = tid; i < BM * HL; i += 256) {
    const int m = i >> 10;
    sProb[i] = (__bf16)(__expf(sS[i] - sRowMax[m]) * sRowRs[m]);
  }
  __syncthreads();

  // ---- Phase D: O(16x256) = Prob(16x1024) @ H(1024x256) -> aligned_hypothesis
#pragma unroll
  for (int t2 = 0; t2 < 2; ++t2) {
    const int nt = wave * 2 + t2;           // 16 n-tiles over d
    v8f acc = {};
    for (int kt = 0; kt < 32; ++kt) {       // K = 1024 (q)
      const int kb = kt * 32;
      const __bf16* arow = sProb + lmod * HL + kb + 8 * lhalf;
      v16bf A = cat8(*(const v8bf*)arow, *(const v8bf*)(arow + 16));
      const __bf16* brow = Ht + ((size_t)(b * D_ + nt * 16 + lmod)) * HL + kb + 8 * lhalf;
      v16bf Bf = cat8(*(const v8bf*)brow, *(const v8bf*)(brow + 16));
      acc = wmma_bf16(A, Bf, acc);
    }
#pragma unroll
    for (int r = 0; r < 8; ++r) {
      const int M = r + 8 * lhalf;
      outHyp[((size_t)(b * PL + p0 + M)) * D_ + nt * 16 + lmod] = acc[r];
    }
  }
}

// ---- epilogue: aligned_premise -------------------------------------------

__global__ __launch_bounds__(256) void k3_premise(const float* __restrict__ P,
                                                  const unsigned* __restrict__ keys,
                                                  float* __restrict__ outPrem) {
  __shared__ float sV[HL];
  __shared__ float sP[HL];
  __shared__ float sRed[256];
  const int b = blockIdx.x;
  const int tid = threadIdx.x;

  float lm = -__builtin_inff();
  for (int q = tid; q < HL; q += 256) {
    float v = fdec(keys[b * HL + q]);
    sV[q] = v;
    lm = fmaxf(lm, v);
  }
  sRed[tid] = lm;
  __syncthreads();
  for (int off = 128; off > 0; off >>= 1) {
    if (tid < off) sRed[tid] = fmaxf(sRed[tid], sRed[tid + off]);
    __syncthreads();
  }
  const float mx = sRed[0];
  __syncthreads();

  float ls = 0.f;
  for (int q = tid; q < HL; q += 256) ls += __expf(sV[q] - mx);
  sRed[tid] = ls;
  __syncthreads();
  for (int off = 128; off > 0; off >>= 1) {
    if (tid < off) sRed[tid] += sRed[tid + off];
    __syncthreads();
  }
  const float rs = 1.0f / sRed[0];
  __syncthreads();

  for (int q = tid; q < HL; q += 256) sP[q] = __expf(sV[q] - mx) * rs;
  __syncthreads();

  const int d = tid;  // blockDim == D_
  float acc = 0.f;
  for (int q = 0; q < HL; ++q) acc += sP[q] * P[((size_t)(b * PL + q)) * D_ + d];
  for (int h = 0; h < HL; ++h)
    outPrem[((size_t)(b * HL + h)) * D_ + d] = acc;
}

// ---- launcher ------------------------------------------------------------

extern "C" void kernel_launch(void* const* d_in, const int* in_sizes, int n_in,
                              void* d_out, int out_size, void* d_ws, size_t ws_size,
                              hipStream_t stream) {
  (void)in_sizes; (void)n_in; (void)out_size; (void)ws_size;
  const float* P     = (const float*)d_in[0];
  const float* H     = (const float*)d_in[1];
  const int*   pmask = (const int*)d_in[2];
  const int*   hmask = (const int*)d_in[3];
  const float* W     = (const float*)d_in[4];
  const float* bias  = (const float*)d_in[5];

  float* out     = (float*)d_out;
  float* outPrem = out;                               // [16,1024,256]
  float* outHyp  = out + (size_t)B_ * HL * D_;        // [16,1024,256]

  char* ws = (char*)d_ws;
  unsigned short* wt   = (unsigned short*)ws;                           // 131072 B
  unsigned short* ht   = (unsigned short*)(ws + 131072);                // 8388608 B
  unsigned short* hbf  = (unsigned short*)(ws + 131072 + 8388608);      // 8388608 B
  unsigned*       keys = (unsigned*)(ws + 131072 + 2 * 8388608);        // 65536 B

  k0_wt  <<<dim3(D_), dim3(D_), 0, stream>>>(W, wt);
  k0_ht  <<<dim3(HL / 16, D_ / 16, B_), dim3(256), 0, stream>>>(H, ht);
  k0_hbf <<<dim3((B_ * HL * D_) / 1024), dim3(256), 0, stream>>>(H, hbf);
  k0_init<<<dim3(B_ * HL / 256), dim3(256), 0, stream>>>(keys);
  k2_fused<<<dim3(PL / BM, B_), dim3(256), SMEM_K2, stream>>>(
      P, pmask, hmask, bias, wt, ht, hbf, keys, outHyp);
  k3_premise<<<dim3(B_), dim3(256), 0, stream>>>(P, keys, outPrem);
}
